// DecoderBlockAEM_56246891708557
// MI455X (gfx1250) — matrix operand order repaired and checked by reading, not compiled, not run on hardware
//
#include <hip/hip_runtime.h>
#include <hip/hip_bf16.h>
#include <math.h>

typedef __attribute__((ext_vector_type(16))) __bf16 v16bf;
typedef __attribute__((ext_vector_type(8)))  __bf16 v8bf;
typedef __attribute__((ext_vector_type(8)))  float  v8f;

#define WMMA_BF16(A, B, C) \
  __builtin_amdgcn_wmma_f32_16x16x32_bf16(false, (A), false, (B), (short)0, (C), false, false)

struct Taps { int n; int dy[9]; int dx[9]; int kidx[9]; };

static __device__ inline v8f v8zero() {
  v8f z;
#pragma unroll
  for (int i = 0; i < 8; ++i) z[i] = 0.f;
  return z;
}
static __device__ inline v16bf v16bfzero() {
  v16bf z;
#pragma unroll
  for (int i = 0; i < 16; ++i) z[i] = (__bf16)0.0f;
  return z;
}

// ---------------------------------------------------------------------------
// Implicit-GEMM conv on NHWC bf16 activations.
//   out[b, oy, ox, oc] = act(scale[oc]*sum_{tap,ic} W * in[b, y+dy, x+dx, ic] + shift[oc])
// Weights pre-swizzled into per-lane WMMA A order: [oc][K/32][half][16].
// LDSW: stage the weight matrix in LDS (shared by the block's 8 waves).
// ---------------------------------------------------------------------------
template <int CIN, int MT, bool LDSW>
__global__ __launch_bounds__(256) void ig_conv(
    const __bf16* __restrict__ in, const __bf16* __restrict__ wmat,
    const float* __restrict__ scale, const float* __restrict__ shift,
    float* __restrict__ out32, __bf16* __restrict__ outbf,
    int Bn, int Hin, int Win, int Hg, int Wg, int Ho, int Wo,
    int sy, int sx, int py, int px, Taps taps, int act, int nchw_out)
{
  const int COUT = MT * 16;
  extern __shared__ __align__(32) char smem_raw[];

  const __bf16* wbase = wmat;
  if constexpr (LDSW) {
    __bf16* sw = (__bf16*)smem_raw;
    const int wele = COUT * taps.n * CIN;
    for (int i = threadIdx.x * 8; i < wele; i += blockDim.x * 8)
      *(v8bf*)(sw + i) = *(const v8bf*)(wmat + i);   // ds_store_b128
    __syncthreads();
    wbase = sw;
  }

  const int wid  = blockIdx.x * (blockDim.x >> 5) + (threadIdx.x >> 5);
  const int lane = threadIdx.x & 31;
  const int wt   = Wg >> 4;
  const int xt   = wid % wt;
  const int y    = (wid / wt) % Hg;
  const int b    = wid / (wt * Hg);
  if (b >= Bn) return;

  const int l16  = lane & 15;
  const int half = lane >> 4;
  const int xg   = (xt << 4) + l16;

  v8f acc[MT];
#pragma unroll
  for (int mt = 0; mt < MT; ++mt) acc[mt] = v8zero();
  const v16bf z16 = v16bfzero();

  const int Kt32 = (taps.n * CIN) >> 5;

  for (int tap = 0; tap < taps.n; ++tap) {
    const int dy = taps.dy[tap], dx = taps.dx[tap];
    const int yin = y + dy;
    const int xi  = xg + dx;
    const bool v  = (yin >= 0) && (yin < Hin) && (xi >= 0) && (xi < Win);
    const __bf16* bpix =
        in + ((size_t)(b * Hin + yin) * Win + xi) * CIN + half * 16;
    if (v) __builtin_prefetch(bpix + (size_t)Win * CIN, 0, 1);  // next row

#pragma unroll
    for (int icb = 0; icb < CIN; icb += 32) {
      const v16bf bv = v ? *(const v16bf*)(bpix + icb) : z16;   // 32B aligned
      const int ks32 = (tap * CIN + icb) >> 5;
#pragma unroll
      for (int mt = 0; mt < MT; ++mt) {
        const __bf16* ap =
            wbase + ((size_t)(mt * 16 + l16) * Kt32 + ks32) * 32 + half * 16;
        const v16bf av = *(const v16bf*)ap;                     // 32B aligned
        acc[mt] = WMMA_BF16(av, bv, acc[mt]);
      }
    }
  }

  const int oy = sy * y + py;
  const int ox = sx * xg + px;
#pragma unroll
  for (int mt = 0; mt < MT; ++mt) {
    float vals[8];
#pragma unroll
    for (int r = 0; r < 8; ++r) {
      const int oc = mt * 16 + half * 8 + r;  // D: VGPR r <-> M = r + half*8
      float vv = acc[mt][r] * scale[oc] + shift[oc];
      if (act == 1) vv = vv > 0.f ? vv : 0.f;
      else if (act == 2) vv = (vv > 20.f) ? vv : log1pf(expf(vv));
      vals[r] = vv;
    }
    if (nchw_out) {
#pragma unroll
      for (int r = 0; r < 8; ++r) {
        const int oc = mt * 16 + half * 8 + r;
        const size_t idx = (((size_t)b * COUT + oc) * Ho + oy) * Wo + ox;
        if (out32) out32[idx] = vals[r];
        if (outbf) outbf[idx] = (__bf16)vals[r];
      }
    } else {
      const size_t base =
          (((size_t)b * Ho + oy) * Wo + ox) * COUT + mt * 16 + half * 8;
      if (out32) {
        v8f o;
#pragma unroll
        for (int r = 0; r < 8; ++r) o[r] = vals[r];
        *(v8f*)(out32 + base) = o;            // 32B aligned
      }
      if (outbf) {
        v8bf o;
#pragma unroll
        for (int r = 0; r < 8; ++r) o[r] = (__bf16)vals[r];
        *(v8bf*)(outbf + base) = o;           // 16B aligned
      }
    }
  }
}

// ---------------------------------------------------------------------------
// Fused q/k/v 1x1 projections: one pass over feat (NHWC bf16).
// Concatenated weights [160 oc][K=128] (q:0-15 softplus, k:16-31 softplus,
// v:32-159 linear), staged in LDS (40KB). Outputs NCHW rows for attention.
// ---------------------------------------------------------------------------
__global__ __launch_bounds__(256) void qkv_conv(
    const __bf16* __restrict__ feat, const __bf16* __restrict__ wqkv,
    const float* __restrict__ sc, const float* __restrict__ sh,
    __bf16* __restrict__ Qb, __bf16* __restrict__ Kb, __bf16* __restrict__ Vb,
    int Bn, int HW)
{
  extern __shared__ __align__(32) char smem_raw[];
  __bf16* sw = (__bf16*)smem_raw;
  for (int i = threadIdx.x * 8; i < 160 * 128; i += blockDim.x * 8)
    *(v8bf*)(sw + i) = *(const v8bf*)(wqkv + i);
  __syncthreads();

  const int wid  = blockIdx.x * (blockDim.x >> 5) + (threadIdx.x >> 5);
  const int lane = threadIdx.x & 31;
  const int nt = wid % (HW >> 4);
  const int b  = wid / (HW >> 4);
  if (b >= Bn) return;
  const int l16 = lane & 15, half = lane >> 4;
  const int p = (nt << 4) + l16;

  v8f acc[10];
#pragma unroll
  for (int mt = 0; mt < 10; ++mt) acc[mt] = v8zero();

  const __bf16* bp = feat + ((size_t)b * HW + p) * 128 + half * 16;
#pragma unroll
  for (int icb = 0; icb < 128; icb += 32) {
    const v16bf bv = *(const v16bf*)(bp + icb);
    const int ks32 = icb >> 5;
#pragma unroll
    for (int mt = 0; mt < 10; ++mt) {
      const v16bf av =
          *(const v16bf*)(sw + ((size_t)(mt * 16 + l16) * 4 + ks32) * 32 + half * 16);
      acc[mt] = WMMA_BF16(av, bv, acc[mt]);
    }
  }

#pragma unroll
  for (int mt = 0; mt < 10; ++mt) {
#pragma unroll
    for (int r = 0; r < 8; ++r) {
      const int oc = mt * 16 + half * 8 + r;
      float vv = acc[mt][r] * sc[oc] + sh[oc];
      if (mt < 2) vv = (vv > 20.f) ? vv : log1pf(expf(vv));
      if (mt == 0)      Qb[((size_t)b * 16 + half * 8 + r) * HW + p] = (__bf16)vv;
      else if (mt == 1) Kb[((size_t)b * 16 + half * 8 + r) * HW + p] = (__bf16)vv;
      else Vb[((size_t)b * 128 + (mt - 2) * 16 + half * 8 + r) * HW + p] = (__bf16)vv;
    }
  }
}

// ---------------------------------------------------------------------------
// KV[b,m,c] = sum_n K[b,m,n] * V[b,c,n]  (WMMA over n, atomic fp32 reduce)
// ---------------------------------------------------------------------------
__global__ __launch_bounds__(256) void kv_gemm(
    const __bf16* __restrict__ Kb, const __bf16* __restrict__ Vb,
    float* __restrict__ KV, int Bn, int N, int nchunks)
{
  const int wid  = blockIdx.x * (blockDim.x >> 5) + (threadIdx.x >> 5);
  const int lane = threadIdx.x & 31;
  const int ct = wid % 8;
  const int ch = (wid / 8) % nchunks;
  const int b  = wid / (8 * nchunks);
  if (b >= Bn) return;
  const int chunk = N / nchunks;
  const int n0 = ch * chunk;
  const int l16 = lane & 15, half = lane >> 4;

  const __bf16* krow = Kb + ((size_t)b * 16 + l16) * N;
  const __bf16* vrow = Vb + ((size_t)b * 128 + ct * 16 + l16) * N;

  v8f acc = v8zero();
  for (int nb = n0; nb < n0 + chunk; nb += 32) {
    const v8bf alo = *(const v8bf*)(krow + nb + half * 16);
    const v8bf ahi = *(const v8bf*)(krow + nb + half * 16 + 8);
    const v16bf av = __builtin_shufflevector(alo, ahi, 0, 1, 2, 3, 4, 5, 6, 7,
                                             8, 9, 10, 11, 12, 13, 14, 15);
    const v16bf bv = *(const v16bf*)(vrow + nb + half * 16);
    acc = WMMA_BF16(av, bv, acc);
  }
#pragma unroll
  for (int r = 0; r < 8; ++r) {
    const int m = r + half * 8;
    const int c = ct * 16 + l16;
    atomicAdd(&KV[((size_t)b * 16 + m) * 128 + c], acc[r]);
  }
}

// ---------------------------------------------------------------------------
// out[b,c,n] (NCHW fp32) =
//   (feat_nhwc[b,n,c] + gamma * norm[n] * sum_m Q[b,m,n]*KV[b,m,c]) * s4[c] + t4[c]
// ---------------------------------------------------------------------------
__global__ __launch_bounds__(256) void wv_bn4(
    const __bf16* __restrict__ Qb, const float* __restrict__ KV,
    const float* __restrict__ normv, const float* __restrict__ feat,
    const float* __restrict__ gamma, const float* __restrict__ sc4,
    const float* __restrict__ sh4, float* __restrict__ out, int Bn, int N)
{
  const int wid  = blockIdx.x * (blockDim.x >> 5) + (threadIdx.x >> 5);
  const int lane = threadIdx.x & 31;
  const int nt = wid % (N >> 4);
  const int b  = wid / (N >> 4);
  if (b >= Bn) return;
  const int l16 = lane & 15, half = lane >> 4;
  const int n = (nt << 4) + l16;
  const float g  = gamma[0];
  const float nv = normv[(size_t)b * N + n];

  v16bf bv;  // B rows = m (only K<16 valid)
#pragma unroll
  for (int e = 0; e < 16; ++e)
    bv[e] = half ? (__bf16)0.0f : Qb[((size_t)b * 16 + e) * N + n];

#pragma unroll
  for (int mt = 0; mt < 8; ++mt) {
    v16bf av;  // A[c][m] = KV[b][m][c]
#pragma unroll
    for (int e = 0; e < 16; ++e) {
      const int kof = (e < 8) ? e : (16 + (e - 8));
      const int m = half * 16 + e;
      av[e] = (m < 16) ? (__bf16)KV[((size_t)b * 16 + m) * 128 + mt * 16 + l16]
                       : (__bf16)0.0f;
    }
    v8f acc = v8zero();
    acc = WMMA_BF16(av, bv, acc);

    const v8f f8 = *(const v8f*)(feat + ((size_t)b * N + n) * 128 + mt * 16 + half * 8);
#pragma unroll
    for (int r = 0; r < 8; ++r) {
      const int c = mt * 16 + r + half * 8;
      const float att = f8[r] + g * acc[r] * nv;
      out[((size_t)b * 128 + c) * N + n] = att * sc4[c] + sh4[c];
    }
  }
}

// ---------------------------------------------------------------------------
// Helper kernels
// ---------------------------------------------------------------------------
__global__ __launch_bounds__(256) void nchw_to_nhwc_bf16(
    const float* __restrict__ src, __bf16* __restrict__ dst, int C, int Npix) {
  __shared__ float tile[32][33];
  const int ptiles = Npix >> 5, ctiles = C >> 5;
  const int bx = blockIdx.x % ptiles;
  const int by = (blockIdx.x / ptiles) % ctiles;
  const int b  = blockIdx.x / (ptiles * ctiles);
  const int p0 = bx << 5, c0 = by << 5;
  const int tx = threadIdx.x, ty = threadIdx.y;  // 32 x 8
#pragma unroll
  for (int j = 0; j < 32; j += 8)
    tile[ty + j][tx] = src[(size_t)(b * C + c0 + ty + j) * Npix + p0 + tx];
  __syncthreads();
#pragma unroll
  for (int j = 0; j < 32; j += 8)
    dst[((size_t)b * Npix + p0 + ty + j) * C + c0 + tx] = (__bf16)tile[tx][ty + j];
}

__global__ void zero_f(float* p, int n) {
  const int i = blockIdx.x * blockDim.x + threadIdx.x;
  if (i < n) p[i] = 0.f;
}

__global__ void ksum_k(const __bf16* __restrict__ Kb, float* __restrict__ Ksum, int N) {
  __shared__ float red[256];
  const int bm = blockIdx.x;
  float s = 0.f;
  for (int n = threadIdx.x; n < N; n += 256) s += (float)Kb[(size_t)bm * N + n];
  red[threadIdx.x] = s;
  __syncthreads();
  for (int st = 128; st > 0; st >>= 1) {
    if ((int)threadIdx.x < st) red[threadIdx.x] += red[threadIdx.x + st];
    __syncthreads();
  }
  if (threadIdx.x == 0) Ksum[bm] = red[0] + 1e-6f;
}

__global__ void norm_k(const __bf16* __restrict__ Qb, const float* __restrict__ Ksum,
                       float* __restrict__ normv, int Bn, int N) {
  const size_t gid = (size_t)blockIdx.x * blockDim.x + threadIdx.x;
  if (gid >= (size_t)Bn * N) return;
  const int b = (int)(gid / N);
  const int n = (int)(gid % N);
  float s = 0.f;
#pragma unroll
  for (int m = 0; m < 16; ++m)
    s += (float)Qb[((size_t)b * 16 + m) * N + n] * Ksum[b * 16 + m];
  normv[gid] = 1.f / s;
}

// Repack weights into per-lane WMMA-A order: dst[((oc*(Ktot/32)+kb)*2+half)*16+e]
__global__ void prep_w(const float* __restrict__ src, __bf16* __restrict__ dst,
                       int O, int I, int KHW, Taps t, int deconv) {
  const int Ktot = t.n * I;
  const int total = O * Ktot;
  const int gid = blockIdx.x * blockDim.x + threadIdx.x;
  if (gid >= total) return;
  const int e    = gid & 15;
  const int half = (gid >> 4) & 1;
  const int kb   = (gid >> 5) % (Ktot >> 5);
  const int oc   = gid / Ktot;
  const int K = kb * 32 + half * 16 + e;
  const int tap = K / I, ic = K % I;
  const int k = t.kidx[tap];
  const float v = deconv ? src[((size_t)ic * O + oc) * KHW + k]
                         : src[((size_t)oc * I + ic) * KHW + k];
  dst[gid] = (__bf16)v;
}

// Fold conv bias + BN running stats into per-channel scale/shift (layout in P).
__global__ void prep_params(
    float* __restrict__ P,
    const float* c1b, const float* b1w, const float* b1b, const float* b1m, const float* b1v,
    const float* dcb, const float* b2w, const float* b2b, const float* b2m, const float* b2v,
    const float* c3b, const float* b3w, const float* b3b, const float* b3m, const float* b3v,
    const float* qb, const float* kb, const float* vb,
    const float* b4w, const float* b4b, const float* b4m, const float* b4v)
{
  const int g = blockIdx.x * blockDim.x + threadIdx.x;
  if (g < 64) {
    const float inv = b1w[g] / sqrtf(b1v[g] + 1e-5f);
    P[0 + g] = inv; P[64 + g] = (c1b[g] - b1m[g]) * inv + b1b[g];
  } else if (g < 128) {
    const int c = g - 64;
    const float inv = b2w[c] / sqrtf(b2v[c] + 1e-5f);
    P[128 + c] = inv; P[192 + c] = (dcb[c] - b2m[c]) * inv + b2b[c];
  } else if (g < 256) {
    const int c = g - 128;
    const float inv = b3w[c] / sqrtf(b3v[c] + 1e-5f);
    P[256 + c] = inv; P[384 + c] = (c3b[c] - b3m[c]) * inv + b3b[c];
  } else if (g < 384) {
    const int c = g - 256;
    const float inv = b4w[c] / sqrtf(b4v[c] + 1e-5f);
    P[832 + c] = inv; P[960 + c] = b4b[c] - b4m[c] * inv;
  } else if (g < 400) {        // qkv combined: q 0-15
    const int c = g - 384; P[1088 + c] = 1.f; P[1248 + c] = qb[c];
  } else if (g < 416) {        // k 16-31
    const int c = g - 400; P[1104 + c] = 1.f; P[1264 + c] = kb[c];
  } else if (g < 544) {        // v 32-159
    const int c = g - 416; P[1120 + c] = 1.f; P[1280 + c] = vb[c];
  }
}

// ---------------------------------------------------------------------------
extern "C" void kernel_launch(void* const* d_in, const int* in_sizes, int n_in,
                              void* d_out, int out_size, void* d_ws, size_t ws_size,
                              hipStream_t stream) {
  (void)in_sizes; (void)n_in; (void)out_size; (void)ws_size;
  const float* x        = (const float*)d_in[0];
  const float* conv1_w  = (const float*)d_in[1];
  const float* conv1_b  = (const float*)d_in[2];
  const float* bn1_w = (const float*)d_in[3], *bn1_b = (const float*)d_in[4];
  const float* bn1_m = (const float*)d_in[5], *bn1_v = (const float*)d_in[6];
  const float* deconv_w = (const float*)d_in[7];
  const float* deconv_b = (const float*)d_in[8];
  const float* bn2_w = (const float*)d_in[9],  *bn2_b = (const float*)d_in[10];
  const float* bn2_m = (const float*)d_in[11], *bn2_v = (const float*)d_in[12];
  const float* conv3_w  = (const float*)d_in[13];
  const float* conv3_b  = (const float*)d_in[14];
  const float* bn3_w = (const float*)d_in[15], *bn3_b = (const float*)d_in[16];
  const float* bn3_m = (const float*)d_in[17], *bn3_v = (const float*)d_in[18];
  const float* q_w = (const float*)d_in[19], *q_b = (const float*)d_in[20];
  const float* k_w = (const float*)d_in[21], *k_b = (const float*)d_in[22];
  const float* v_w = (const float*)d_in[23], *v_b = (const float*)d_in[24];
  const float* gamma = (const float*)d_in[25];
  const float* bn4_w = (const float*)d_in[26], *bn4_b = (const float*)d_in[27];
  const float* bn4_m = (const float*)d_in[28], *bn4_v = (const float*)d_in[29];

  const int B = 4, C0 = 256, C4 = 64, CF = 128, Mq = 16;
  const size_t N0 = 128 * 128, N1 = 256 * 256;

  char* ws = (char*)d_ws;
  size_t off = 0;
  auto take = [&](size_t bytes) -> char* {
    char* p = ws + off;
    off = (off + bytes + 255) & ~(size_t)255;
    return p;
  };

  __bf16* xbf   = (__bf16*)take((size_t)B * C0 * N0 * 2);  // NHWC
  __bf16* y1bf  = (__bf16*)take((size_t)B * C4 * N0 * 2);  // NHWC
  __bf16* y2bf  = (__bf16*)take((size_t)B * C4 * N1 * 2);  // NHWC
  float*  featf = (float*)take((size_t)B * CF * N1 * 4);   // NHWC
  __bf16* featbf= (__bf16*)take((size_t)B * CF * N1 * 2);  // NHWC
  __bf16* Qbf   = (__bf16*)take((size_t)B * Mq * N1 * 2);  // NCHW
  __bf16* Kbf   = (__bf16*)take((size_t)B * Mq * N1 * 2);  // NCHW
  __bf16* Vbf   = (__bf16*)take((size_t)B * CF * N1 * 2);  // NCHW
  float*  KV    = (float*)take((size_t)B * Mq * CF * 4);
  float*  Ksum  = (float*)take((size_t)B * Mq * 4);
  float*  normv = (float*)take((size_t)B * N1 * 4);
  __bf16* Wc1   = (__bf16*)take(256 * 64 * 2);
  __bf16* Wc3   = (__bf16*)take(576 * 128 * 2);
  __bf16* Wqkv  = (__bf16*)take(160 * 128 * 2);
  __bf16* Wd[4];
  for (int p = 0; p < 4; ++p) Wd[p] = (__bf16*)take(4 * 64 * 64 * 2);
  float* P = (float*)take(1408 * 4);

  // ---- parameter / weight prep ----
  prep_params<<<3, 256, 0, stream>>>(P,
      conv1_b, bn1_w, bn1_b, bn1_m, bn1_v,
      deconv_b, bn2_w, bn2_b, bn2_m, bn2_v,
      conv3_b, bn3_w, bn3_b, bn3_m, bn3_v,
      q_b, k_b, v_b, bn4_w, bn4_b, bn4_m, bn4_v);

  Taps t1{}; t1.n = 1; t1.dy[0] = 0; t1.dx[0] = 0; t1.kidx[0] = 0;
  Taps t3{}; t3.n = 9;
  for (int t = 0; t < 9; ++t) { t3.dy[t] = t / 3 - 1; t3.dx[t] = t % 3 - 1; t3.kidx[t] = t; }

  prep_w<<<64, 256, 0, stream>>>(conv1_w, Wc1, 64, 256, 1, t1, 0);
  prep_w<<<288, 256, 0, stream>>>(conv3_w, Wc3, 128, 64, 9, t3, 0);
  // concatenated q/k/v weight matrix (oc-major swizzled rows are independent)
  prep_w<<<8, 256, 0, stream>>>(q_w, Wqkv, 16, 128, 1, t1, 0);
  prep_w<<<8, 256, 0, stream>>>(k_w, Wqkv + 16 * 128, 16, 128, 1, t1, 0);
  prep_w<<<64, 256, 0, stream>>>(v_w, Wqkv + 32 * 128, 128, 128, 1, t1, 0);

  // deconv s2/k3/p1/op1 -> 4 sub-pixel parity GEMMs
  Taps tp[4];
  for (int py = 0; py < 2; ++py) for (int px = 0; px < 2; ++px) {
    const int p = py * 2 + px;
    const int ny = py ? 2 : 1, nx = px ? 2 : 1;
    Taps& tt = tp[p]; tt = Taps{}; tt.n = ny * nx;
    int idx = 0;
    for (int iy = 0; iy < ny; ++iy) for (int ix = 0; ix < nx; ++ix) {
      const int kyv = py ? (iy == 0 ? 0 : 2) : 1;
      const int dyv = py ? (iy == 0 ? 0 : 1) : 0;
      const int kxv = px ? (ix == 0 ? 0 : 2) : 1;
      const int dxv = px ? (ix == 0 ? 0 : 1) : 0;
      tt.dy[idx] = dyv; tt.dx[idx] = dxv;
      tt.kidx[idx] = (2 - kyv) * 3 + (2 - kxv);
      ++idx;
    }
    const int tot = tt.n * 64 * 64;
    prep_w<<<(tot + 255) / 256, 256, 0, stream>>>(deconv_w, Wd[p], 64, 64, 9, tt, 1);
  }

  // ---- pipeline ----
  nchw_to_nhwc_bf16<<<dim3(16384), dim3(32, 8), 0, stream>>>(x, xbf, C0, (int)N0);

  // conv1 + bn1 + relu (weights staged in 32KB LDS)
  ig_conv<256, 4, true><<<512, 256, 64 * 256 * 2, stream>>>(
      xbf, Wc1, P + 0, P + 64, nullptr, y1bf,
      B, 128, 128, 128, 128, 128, 128, 1, 1, 0, 0, t1, 1, 0);

  // deconv + bn2 + relu : 4 parity GEMMs (weights staged in LDS)
  for (int py = 0; py < 2; ++py) for (int px = 0; px < 2; ++px) {
    const int p = py * 2 + px;
    ig_conv<64, 4, true><<<512, 256, (size_t)tp[p].n * 64 * 64 * 2, stream>>>(
        y1bf, Wd[p], P + 128, P + 192, nullptr, y2bf,
        B, 128, 128, 128, 128, 256, 256, 2, 2, py, px, tp[p], 1, 0);
  }

  // conv3 (3x3, pad1) + bn3 + relu -> feat NHWC fp32 + bf16 (weights via L2)
  ig_conv<64, 8, false><<<2048, 256, 0, stream>>>(
      y2bf, Wc3, P + 256, P + 384, featf, featbf,
      B, 256, 256, 256, 256, 256, 256, 1, 1, 0, 0, t3, 1, 0);

  // fused q/k/v projections: single pass over featbf, weights staged in 40KB LDS
  qkv_conv<<<2048, 256, 160 * 128 * 2, stream>>>(
      featbf, Wqkv, P + 1088, P + 1248, Qbf, Kbf, Vbf, B, (int)N1);

  // attention tail
  zero_f<<<32, 256, 0, stream>>>(KV, B * Mq * CF);
  ksum_k<<<64, 256, 0, stream>>>(Kbf, Ksum, (int)N1);
  kv_gemm<<<256, 256, 0, stream>>>(Kbf, Vbf, KV, B, (int)N1, 64);
  norm_k<<<1024, 256, 0, stream>>>(Qbf, Ksum, normv, B, (int)N1);
  wv_bn4<<<2048, 256, 0, stream>>>(Qbf, KV, normv, featf, gamma, P + 832, P + 960,
                                   (float*)d_out, B, (int)N1);
}
